// UMT5Attention_24266565222476
// MI455X (gfx1250) — matrix operand order repaired and checked by reading, not compiled, not run on hardware
//
#include <hip/hip_runtime.h>
#include <cmath>

#define DEV __device__ __forceinline__

typedef __attribute__((ext_vector_type(16))) __bf16 v16bf;
typedef __attribute__((ext_vector_type(8)))  __bf16 v8bf;
typedef __attribute__((ext_vector_type(8)))  float  v8f;

constexpr int T    = 3072;
constexpr int D    = 768;
constexpr int H    = 12;
constexpr int DK   = 64;
constexpr int NTAB = 2 * T - 1;   // bias table over delta = t_k - t_q

// ---------------------------------------------------------------- helpers

DEV __bf16 f2bf(float f) {
  unsigned u = __builtin_bit_cast(unsigned, f);
  unsigned r = u + 0x7fffu + ((u >> 16) & 1u);     // round-to-nearest-even
  return __builtin_bit_cast(__bf16, (unsigned short)(r >> 16));
}

DEV v8f vzero8() {
  v8f z;
#pragma unroll
  for (int i = 0; i < 8; ++i) z[i] = 0.0f;
  return z;
}

DEV v8f wmma_bf16(v16bf a, v16bf b, v8f c) {
  // (neg_a, A, neg_b, B, c_mod, C, reuse_a, reuse_b)
  return __builtin_amdgcn_wmma_f32_16x16x32_bf16(false, a, false, b, (short)0, c,
                                                 false, false);
}

// A fragment (16x32 bf16): lane&15 = row M; grp = lane>>4:
//   elems i -> K = k0 + grp*16 + i
DEV v16bf load_a_frag(const __bf16* rowp_k0, int grp) {
  return *(const v16bf*)(rowp_k0 + grp * 16);
}

// B fragment (32x16 bf16): lane&15 = column N; grp = lane>>4:
//   elems i -> K = k0 + grp*16 + i  => one v16bf load from [N][K] layout.

// ---------------------------------------------------------------- setup kernels

__global__ void cast_bf16_kernel(const float* __restrict__ in,
                                 __bf16* __restrict__ out, int n) {
  int i = blockIdx.x * blockDim.x + threadIdx.x;
  if (i < n) out[i] = f2bf(in[i]);
}

// out[n*D + k] = in[k*D + n]  (store W^T in bf16 so B-fragments are contiguous)
__global__ void transpose_bf16_kernel(const float* __restrict__ in,
                                      __bf16* __restrict__ out) {
  int i = blockIdx.x * blockDim.x + threadIdx.x;
  if (i >= D * D) return;
  int k = i / D, n = i - k * D;
  out[n * D + k] = f2bf(in[i]);
}

__global__ void seg_kernel(const int* __restrict__ seq_lens, int nseq,
                           int* __restrict__ seg) {
  int t = blockIdx.x * blockDim.x + threadIdx.x;
  if (t >= T) return;
  int c = 0, s = 0;
  for (int i = 0; i < nseq; ++i) {
    int b = c; c += seq_lens[i];
    if (t >= b && t < c) s = i;
  }
  seg[t] = s;
}

// T5 relative-position bucket bias, tabulated over delta = pos_k - pos_q.
// (Within a segment pos_k - pos_q == t_k - t_q; cross-segment pairs are masked.)
__global__ void bias_table_kernel(const float* __restrict__ rel_bias,
                                  float* __restrict__ tab) {
  int i = blockIdx.x * blockDim.x + threadIdx.x;
  if (i >= NTAB) return;
  int delta = i - (T - 1);
  int n = -delta;                       // reference: n = -rel_pos
  int ret = (n < 0) ? 16 : 0;           // NUM_BUCKETS/2 = 16
  n = (n < 0) ? -n : n;
  int b;
  if (n < 8) {                          // max_exact = 8
    b = n;
  } else {
    int v = 8 + (int)(logf((float)n * (1.0f / 8.0f)) * (8.0f / logf(16.0f)));
    b = (v < 15) ? v : 15;
  }
  b += ret;
  for (int h = 0; h < H; ++h) tab[h * NTAB + i] = rel_bias[b * H + h];
}

// ---------------------------------------------------------------- bf16 WMMA GEMM
// C[M,N] = A[M,K] * B[K,N], A bf16 row-major, Bt = B^T bf16 ([N][K] row-major).
// Block: 256 threads = 8 waves (4x2), wave tile 32x64, block tile 128x128.
// OUT_MODE: 0 = bf16 row-major [M][N], 1 = bf16 transposed [N][M], 2 = f32 [M][N].
template <int OUT_MODE>
__global__ __launch_bounds__(256) void gemm_bf16_kernel(
    const __bf16* __restrict__ A, const __bf16* __restrict__ Bt,
    void* __restrict__ Cout, int M, int N, int K) {
  const int lane = threadIdx.x & 31;
  const int wave = threadIdx.x >> 5;
  const int wm = wave >> 1, wn = wave & 1;
  const int grp = lane >> 4, l16 = lane & 15;
  const int m0 = blockIdx.x * 128 + wm * 32;
  const int n0 = blockIdx.y * 128 + wn * 64;

  v8f acc[2][4];
#pragma unroll
  for (int i = 0; i < 2; ++i)
#pragma unroll
    for (int j = 0; j < 4; ++j) acc[i][j] = vzero8();

  const __bf16* arow = A + (size_t)(m0 + l16) * K;
  const __bf16* bcol = Bt + (size_t)(n0 + l16) * K;

  for (int k0 = 0; k0 < K; k0 += 32) {
    v16bf af[2], bfr[4];
    af[0] = load_a_frag(arow + k0, grp);
    af[1] = load_a_frag(arow + (size_t)16 * K + k0, grp);
#pragma unroll
    for (int j = 0; j < 4; ++j)
      bfr[j] = *(const v16bf*)(bcol + (size_t)j * 16 * K + k0 + grp * 16);
#pragma unroll
    for (int i = 0; i < 2; ++i)
#pragma unroll
      for (int j = 0; j < 4; ++j) acc[i][j] = wmma_bf16(af[i], bfr[j], acc[i][j]);
  }

  // C layout: lane l holds N = n_tile + l16, rows M = m_tile + grp*8 + r.
#pragma unroll
  for (int i = 0; i < 2; ++i) {
#pragma unroll
    for (int j = 0; j < 4; ++j) {
      int mb = m0 + i * 16 + grp * 8;
      int n = n0 + j * 16 + l16;
#pragma unroll
      for (int r = 0; r < 8; ++r) {
        int m = mb + r;
        if (OUT_MODE == 0) ((__bf16*)Cout)[(size_t)m * N + n] = f2bf(acc[i][j][r]);
        else if (OUT_MODE == 1) ((__bf16*)Cout)[(size_t)n * M + m] = f2bf(acc[i][j][r]);
        else ((float*)Cout)[(size_t)m * N + n] = acc[i][j][r];
      }
    }
  }
}

// ---------------------------------------------------------------- flash attention
// Grid: (T/128 q-tiles, H heads). Block: 128 threads = 4 waves; each wave owns
// 32 q rows (two 16-row m-fragments sharing every K and V fragment -> 2x the
// arithmetic intensity of a 16-row wave). Qb,Kb: bf16 [T][768]. Vt: bf16 [H*64][T].
__global__ __launch_bounds__(128) void attn_kernel(
    const __bf16* __restrict__ Qb, const __bf16* __restrict__ Kb,
    const __bf16* __restrict__ Vt, const int* __restrict__ seg,
    const float* __restrict__ bias_tab, const int* __restrict__ seq_lens,
    int nseq, __bf16* __restrict__ attn_out) {
  __shared__ __bf16 lds_p[4 * 2 * 16 * 64];  // per (wave, m-frag) 16x64 bf16 P

  const int lane = threadIdx.x & 31;
  const int wave = threadIdx.x >> 5;
  const int grp = lane >> 4, l16 = lane & 15;
  const int h = blockIdx.y;
  const int qb = blockIdx.x * 128;
  const int q0 = qb + wave * 32;

  // k range = union of segments covering this q-tile, rounded to 64.
  int kstart = 0, kend = T;
  {
    int c = 0;
    for (int s = 0; s < nseq; ++s) {
      int b = c; c += seq_lens[s];
      if (qb >= b && qb < c) kstart = b;
      if (qb + 127 >= b && qb + 127 < c) kend = c;
    }
    kstart &= ~63;
    kend = (kend + 63) & ~63;
  }

  // Q A-fragments (two m-frags x two K=32 chunks), resident for whole kernel.
  v16bf qa[2][2];
#pragma unroll
  for (int i = 0; i < 2; ++i) {
    const __bf16* qrow = Qb + (size_t)(q0 + i * 16 + l16) * D + h * DK;
    qa[i][0] = load_a_frag(qrow, grp);
    qa[i][1] = load_a_frag(qrow + 32, grp);
  }

  float m_run[2][8], lsum[2][8];
  v8f o[2][4];
  int qtok[2][8], segq[2][8];
#pragma unroll
  for (int i = 0; i < 2; ++i)
#pragma unroll
    for (int r = 0; r < 8; ++r) {
      m_run[i][r] = -3.0e38f;
      lsum[i][r] = 0.0f;
      int tq = q0 + i * 16 + grp * 8 + r;
      qtok[i][r] = tq;
      segq[i][r] = seg[tq];
    }
#pragma unroll
  for (int i = 0; i < 2; ++i)
#pragma unroll
    for (int j = 0; j < 4; ++j) o[i][j] = vzero8();

  const float* btab = bias_tab + (size_t)h * NTAB + (T - 1);

  for (int kt = kstart; kt < kend; kt += 64) {
    // ---- S = Q K^T (32 x 64): K frags shared by both m-frags, 16 WMMAs
    v8f s[2][4];
#pragma unroll
    for (int i = 0; i < 2; ++i)
#pragma unroll
      for (int j = 0; j < 4; ++j) s[i][j] = vzero8();
#pragma unroll
    for (int c = 0; c < 2; ++c) {
      v16bf kb[4];
#pragma unroll
      for (int j = 0; j < 4; ++j)
        kb[j] = *(const v16bf*)(Kb + (size_t)(kt + j * 16 + l16) * D + h * DK +
                                c * 32 + grp * 16);
#pragma unroll
      for (int i = 0; i < 2; ++i)
#pragma unroll
        for (int j = 0; j < 4; ++j) s[i][j] = wmma_bf16(qa[i][c], kb[j], s[i][j]);
    }

    // ---- bias + segment mask (delta = t_k - t_q is exact within a segment)
    int ktok[4], segk[4];
#pragma unroll
    for (int j = 0; j < 4; ++j) {
      ktok[j] = kt + j * 16 + l16;
      segk[j] = seg[ktok[j]];
    }
#pragma unroll
    for (int i = 0; i < 2; ++i)
#pragma unroll
      for (int j = 0; j < 4; ++j)
#pragma unroll
        for (int r = 0; r < 8; ++r) {
          float v = s[i][j][r] + btab[ktok[j] - qtok[i][r]];
          if (segk[j] != segq[i][r]) v = -1.0e30f;
          s[i][j][r] = v;
        }

    // ---- online softmax + P -> LDS
#pragma unroll
    for (int i = 0; i < 2; ++i) {
      __bf16* myp = lds_p + (size_t)(wave * 2 + i) * 16 * 64;
#pragma unroll
      for (int r = 0; r < 8; ++r) {
        float mr = fmaxf(fmaxf(s[i][0][r], s[i][1][r]),
                         fmaxf(s[i][2][r], s[i][3][r]));
#pragma unroll
        for (int off = 1; off < 16; off <<= 1)
          mr = fmaxf(mr, __shfl_xor(mr, off, 32));
        float mnew = fmaxf(m_run[i][r], mr);
        float scale = __expf(m_run[i][r] - mnew);
        m_run[i][r] = mnew;
        lsum[i][r] *= scale;
#pragma unroll
        for (int j = 0; j < 4; ++j) o[i][j][r] *= scale;
        float rs = 0.0f;
#pragma unroll
        for (int j = 0; j < 4; ++j) {
          float p = __expf(s[i][j][r] - mnew);
          rs += p;
          s[i][j][r] = p;
        }
        lsum[i][r] += rs;
      }
#pragma unroll
      for (int j = 0; j < 4; ++j)
#pragma unroll
        for (int r = 0; r < 8; ++r)
          myp[(grp * 8 + r) * 64 + j * 16 + l16] = f2bf(s[i][j][r]);
    }

    // ---- O += P V (32 x 64): V frags shared by both m-frags, 16 WMMAs
#pragma unroll
    for (int c = 0; c < 2; ++c) {
      v16bf pa[2];
#pragma unroll
      for (int i = 0; i < 2; ++i)
        pa[i] = load_a_frag(lds_p + (size_t)(wave * 2 + i) * 16 * 64 +
                            l16 * 64 + c * 32, grp);
#pragma unroll
      for (int j = 0; j < 4; ++j) {
        v16bf vb = *(const v16bf*)(Vt + (size_t)(h * DK + j * 16 + l16) * T +
                                   kt + c * 32 + grp * 16);
        o[0][j] = wmma_bf16(pa[0], vb, o[0][j]);
        o[1][j] = wmma_bf16(pa[1], vb, o[1][j]);
      }
    }
  }

  // ---- epilogue: normalize and store bf16 [T][768]
#pragma unroll
  for (int i = 0; i < 2; ++i) {
#pragma unroll
    for (int r = 0; r < 8; ++r) {
      float l = lsum[i][r];
#pragma unroll
      for (int off = 1; off < 16; off <<= 1) l += __shfl_xor(l, off, 32);
      float inv = 1.0f / fmaxf(l, 1e-20f);
#pragma unroll
      for (int j = 0; j < 4; ++j) o[i][j][r] *= inv;
    }
#pragma unroll
    for (int j = 0; j < 4; ++j)
#pragma unroll
      for (int r = 0; r < 8; ++r)
        attn_out[(size_t)(q0 + i * 16 + grp * 8 + r) * D + h * DK + j * 16 +
                 l16] = f2bf(o[i][j][r]);
  }
}

// ---------------------------------------------------------------- launch

extern "C" void kernel_launch(void* const* d_in, const int* in_sizes, int n_in,
                              void* d_out, int out_size, void* d_ws,
                              size_t ws_size, hipStream_t stream) {
  (void)n_in; (void)out_size; (void)ws_size;
  const float* x        = (const float*)d_in[0];
  const int*   seq_lens = (const int*)d_in[1];
  const float* Wq       = (const float*)d_in[2];
  const float* Wk       = (const float*)d_in[3];
  const float* Wv       = (const float*)d_in[4];
  const float* Wo       = (const float*)d_in[5];
  const float* rel_bias = (const float*)d_in[6];
  const int nseq = in_sizes[1];

  char* ws = (char*)d_ws;
  size_t off = 0;
  auto carve = [&](size_t bytes) {
    char* p = ws + off;
    off = (off + bytes + 255) & ~(size_t)255;
    return p;
  };
  __bf16* xb   = (__bf16*)carve((size_t)T * D * 2);
  __bf16* WqT  = (__bf16*)carve((size_t)D * D * 2);
  __bf16* WkT  = (__bf16*)carve((size_t)D * D * 2);
  __bf16* WvT  = (__bf16*)carve((size_t)D * D * 2);
  __bf16* WoT  = (__bf16*)carve((size_t)D * D * 2);
  __bf16* Qb   = (__bf16*)carve((size_t)T * D * 2);
  __bf16* Kb   = (__bf16*)carve((size_t)T * D * 2);
  __bf16* Vt   = (__bf16*)carve((size_t)T * D * 2);  // [H*64][T]
  __bf16* attn = (__bf16*)carve((size_t)T * D * 2);
  int*    seg  = (int*)carve((size_t)T * 4);
  float*  btab = (float*)carve((size_t)H * NTAB * 4);

  // setup
  cast_bf16_kernel<<<(T * D + 255) / 256, 256, 0, stream>>>(x, xb, T * D);
  transpose_bf16_kernel<<<(D * D + 255) / 256, 256, 0, stream>>>(Wq, WqT);
  transpose_bf16_kernel<<<(D * D + 255) / 256, 256, 0, stream>>>(Wk, WkT);
  transpose_bf16_kernel<<<(D * D + 255) / 256, 256, 0, stream>>>(Wv, WvT);
  transpose_bf16_kernel<<<(D * D + 255) / 256, 256, 0, stream>>>(Wo, WoT);
  seg_kernel<<<(T + 255) / 256, 256, 0, stream>>>(seq_lens, nseq, seg);
  bias_table_kernel<<<(NTAB + 255) / 256, 256, 0, stream>>>(rel_bias, btab);

  // projections: Q, K natural bf16; V stored transposed [H*64][T]
  dim3 gg(T / 128, D / 128);
  gemm_bf16_kernel<0><<<gg, 256, 0, stream>>>(xb, WqT, Qb, T, D, D);
  gemm_bf16_kernel<0><<<gg, 256, 0, stream>>>(xb, WkT, Kb, T, D, D);
  gemm_bf16_kernel<1><<<gg, 256, 0, stream>>>(xb, WvT, Vt, T, D, D);

  // flash attention
  attn_kernel<<<dim3(T / 128, H), 128, 0, stream>>>(Qb, Kb, Vt, seg, btab,
                                                    seq_lens, nseq, attn);

  // output projection -> fp32 d_out
  gemm_bf16_kernel<2><<<gg, 256, 0, stream>>>(attn, WoT, (float*)d_out, T, D, D);
}
